// TinySSM1D_46574625358437
// MI455X (gfx1250) — matrix operand order repaired and checked by reading, not compiled, not run on hardware
//
#include <hip/hip_runtime.h>
#include <hip/hip_bf16.h>
#include <stdint.h>

#define B_DIM 4
#define L_DIM 8192
#define D_DIM 1024
#define M_TOT (B_DIM * L_DIM)   // 32768 rows

#define MTILE 128
#define NTILE 64
#define KTILE 32
#define NKSTEPS (D_DIM / KTILE)  // 32
#define LDA 40                   // padded LDS row stride (halfs)
#define LDB 40

#define CHUNK 128
#define NCH (L_DIM / CHUNK)      // 64 chunks

typedef __attribute__((ext_vector_type(16))) __bf16    v16bf;
typedef __attribute__((ext_vector_type(8)))  float     v8f;
typedef __attribute__((ext_vector_type(8)))  unsigned  v8u;

__device__ __forceinline__ unsigned short f2bf(float f) {
    unsigned u = __builtin_bit_cast(unsigned, f);
    u += 0x7FFFu + ((u >> 16) & 1u);          // round-to-nearest-even
    return (unsigned short)(u >> 16);
}

__device__ __forceinline__ float sigmoidf_(float x) {
    return 1.0f / (1.0f + __expf(-x));
}

__device__ __forceinline__ v16bf make_frag(uint4 lo, uint4 hi) {
    v8u u;
    u[0] = lo.x; u[1] = lo.y; u[2] = lo.z; u[3] = lo.w;
    u[4] = hi.x; u[5] = hi.y; u[6] = hi.z; u[7] = hi.w;
    return __builtin_bit_cast(v16bf, u);
}

// Async 16B global -> LDS copy (CDNA5 copy engine, tracked by ASYNCcnt).
// lds_off: raw LDS byte offset (low 32 bits of the generic LDS address).
__device__ __forceinline__ void async_copy16(unsigned lds_off, const void* gptr) {
    asm volatile("global_load_async_to_lds_b128 %0, %1, off"
                 :: "v"(lds_off), "v"(gptr) : "memory");
}

template <int N>
__device__ __forceinline__ void wait_async() {
    asm volatile("s_wait_asynccnt %0" :: "n"(N) : "memory");
}

__device__ __forceinline__ unsigned lds_off(const void* p) {
    return (unsigned)(size_t)p;   // LDS generic addr: bits[31:0] = LDS offset
}

// ---------------- f32 -> bf16 cast (packed, coalesced) ----------------
__global__ __launch_bounds__(256)
void cvt2_kernel(const float2* __restrict__ in, unsigned* __restrict__ out, int n2) {
    int i = blockIdx.x * 256 + threadIdx.x;
    int stride = gridDim.x * 256;
    for (; i < n2; i += stride) {
        float2 f = in[i];
        out[i] = (unsigned)f2bf(f.x) | ((unsigned)f2bf(f.y) << 16);
    }
}

// ---------------- bf16 WMMA GEMM: C[M,N] = A[M,K] * W[N,K]^T ----------------
// NPROJ==3: W is [3D, D]; epilogue fuses activations, writes v = sig(i)*tanh(u)
//           and o = sig(o) as f32 into out_v / out_o (each [M_TOT, D]).
// NPROJ==1: plain GEMM, f32 result to out_v.
// Double-buffered LDS; tiles staged with global_load_async_to_lds_b128.
template <int NPROJ>
__global__ __launch_bounds__(256)
void gemm_bf16_kernel(const unsigned short* __restrict__ Abuf,   // [M_TOT, D] bf16
                      const unsigned short* __restrict__ Wbuf,   // [NPROJ*D, D] bf16
                      float* __restrict__ out_v,
                      float* __restrict__ out_o) {
    static_assert(NTILE == 64 && KTILE == 32 && MTILE == 128, "tile config");
    constexpr int PEND = 2 + NPROJ;   // async instructions issued per stage per wave

    __shared__ unsigned short sA[2][MTILE * LDA];
    __shared__ unsigned short sB[2][NPROJ][NTILE * LDB];

    const int tid   = threadIdx.x;
    const int lane  = tid & 31;
    const int w     = tid >> 5;
    const int wm    = w & 3;      // wave M quadrant (32 rows each)
    const int wn    = w >> 2;     // wave N half (32 cols each)
    const int lrow  = lane & 15;
    const int lhalf = lane >> 4;

    const int m0 = blockIdx.x * MTILE;
    const int n0 = blockIdx.y * NTILE;

    v8f zero = {};
    v8f acc[NPROJ][2][2];
#pragma unroll
    for (int p = 0; p < NPROJ; ++p)
#pragma unroll
        for (int mt = 0; mt < 2; ++mt)
#pragma unroll
            for (int nt = 0; nt < 2; ++nt) acc[p][mt][nt] = zero;

    // staging coordinates (one thread = fixed chunks)
    const int arow  = tid >> 1;          // 2 threads per A row
    const int acolh = (tid & 1) * 16;    // half-offset within 32-wide K slice
    const int brow  = tid >> 2;          // 4 threads per B row
    const int bq    = (tid & 3) * 8;     // 8-half (16B) chunk within row

    auto stage = [&](int buf, int k0) {
        const unsigned short* ga = Abuf + (size_t)(m0 + arow) * D_DIM + k0 + acolh;
        if (k0 + KTILE < D_DIM) __builtin_prefetch(ga + KTILE, 0, 1);  // global_prefetch_b8
        async_copy16(lds_off(&sA[buf][arow * LDA + acolh]), ga);
        async_copy16(lds_off(&sA[buf][arow * LDA + acolh + 8]), ga + 8);
#pragma unroll
        for (int p = 0; p < NPROJ; ++p) {
            const unsigned short* gb =
                Wbuf + (size_t)(p * D_DIM + n0 + brow) * D_DIM + k0 + bq;
            async_copy16(lds_off(&sB[buf][p][brow * LDB + bq]), gb);
        }
    };

    stage(0, 0);

    for (int kt = 0; kt < NKSTEPS; ++kt) {
        const int cb = kt & 1;
        if (kt + 1 < NKSTEPS) {
            stage(cb ^ 1, (kt + 1) * KTILE);   // fill other buffer while computing
            wait_async<PEND>();                // retire current buffer's fills only
        } else {
            wait_async<0>();
        }
        __syncthreads();

        // A fragments: lane<16 -> K 0-7 & 16-23 ; lane>=16 -> K 8-15 & 24-31
        v16bf afr[2];
#pragma unroll
        for (int mt = 0; mt < 2; ++mt) {
            const unsigned short* ap = &sA[cb][(wm * 32 + mt * 16 + lrow) * LDA];
            uint4 lo = *(const uint4*)(ap + lhalf * 8);
            uint4 hi = *(const uint4*)(ap + 16 + lhalf * 8);
            afr[mt] = make_frag(lo, hi);
        }
#pragma unroll
        for (int p = 0; p < NPROJ; ++p) {
#pragma unroll
            for (int nt = 0; nt < 2; ++nt) {
                // B fragment: lane = N column; lane<16 -> K 0-7 & 16-23 ; lane>=16 -> K 8-15 & 24-31
                const unsigned short* bp =
                    &sB[cb][p][(wn * 32 + nt * 16 + lrow) * LDB];
                uint4 lo = *(const uint4*)(bp + lhalf * 8);
                uint4 hi = *(const uint4*)(bp + 16 + lhalf * 8);
                v16bf bfr = make_frag(lo, hi);
#pragma unroll
                for (int mt = 0; mt < 2; ++mt) {
                    acc[p][mt][nt] = __builtin_amdgcn_wmma_f32_16x16x32_bf16(
                        false, afr[mt], false, bfr, (short)0, acc[p][mt][nt], false, false);
                }
            }
        }
        __syncthreads();   // all waves done reading buffer cb before it is refilled
    }

    // epilogue: C layout -> VGPR j holds (M = j + 8*lhalf, N = lrow)
    const int colb = n0 + wn * 32 + lrow;
#pragma unroll
    for (int mt = 0; mt < 2; ++mt) {
#pragma unroll
        for (int nt = 0; nt < 2; ++nt) {
            const int col   = colb + nt * 16;
            const int rbase = m0 + wm * 32 + mt * 16 + lhalf * 8;
            if (NPROJ == 3) {
                v8f u  = acc[0][mt][nt];
                v8f ii = acc[1][mt][nt];
                v8f oo = acc[2][mt][nt];
#pragma unroll
                for (int j = 0; j < 8; ++j) {
                    size_t idx = (size_t)(rbase + j) * D_DIM + col;
                    out_v[idx] = sigmoidf_(ii[j]) * tanhf(u[j]);
                    out_o[idx] = sigmoidf_(oo[j]);
                }
            } else {
                v8f u = acc[0][mt][nt];
#pragma unroll
                for (int j = 0; j < 8; ++j)
                    out_v[(size_t)(rbase + j) * D_DIM + col] = u[j];
            }
        }
    }
}

// ---------------- chunked linear-recurrence scan ----------------
// pass1: per-chunk local scan with s0 = 0; emit chunk-end state r[c,b,d]
__global__ __launch_bounds__(256)
void scan_pass1(const float* __restrict__ v, const float* __restrict__ Avec,
                float* __restrict__ r) {
    int idx = blockIdx.x * 256 + threadIdx.x;     // [0, NCH*B*D)
    int d = idx & (D_DIM - 1);
    int b = (idx >> 10) & (B_DIM - 1);
    int c = idx >> 12;
    float a = fminf(__expf(Avec[d]), 0.999f);
    size_t base = (size_t)(b * L_DIM + c * CHUNK) * D_DIM + d;
    float s = 0.0f;
    for (int l = 0; l < CHUNK; ++l)
        s = fmaf(a, s, v[base + (size_t)l * D_DIM]);
    r[(size_t)c * (B_DIM * D_DIM) + (b << 10) + d] = s;
}

// pass2: sequential prefix over the 64 chunk summaries -> carry-in per chunk
__global__ __launch_bounds__(256)
void scan_pass2(const float* __restrict__ r, const float* __restrict__ Avec,
                float* __restrict__ carry) {
    int idx = blockIdx.x * 256 + threadIdx.x;     // [0, B*D)
    int d = idx & (D_DIM - 1);
    float a = fminf(__expf(Avec[d]), 0.999f);
    float p = a;
#pragma unroll
    for (int t = 0; t < 7; ++t) p = p * p;        // a^128 = a^CHUNK
    float s = 0.0f;
    for (int c = 0; c < NCH; ++c) {
        size_t o = (size_t)c * (B_DIM * D_DIM) + idx;
        carry[o] = s;
        s = fmaf(p, s, r[o]);
    }
}

// pass3: replay chunk from its carry, y = o * s, emit bf16 for GEMM2
__global__ __launch_bounds__(256)
void scan_pass3(const float* __restrict__ v, const float* __restrict__ o,
                const float* __restrict__ Avec, const float* __restrict__ carry,
                unsigned short* __restrict__ yb) {
    int idx = blockIdx.x * 256 + threadIdx.x;
    int d = idx & (D_DIM - 1);
    int b = (idx >> 10) & (B_DIM - 1);
    int c = idx >> 12;
    float a = fminf(__expf(Avec[d]), 0.999f);
    float s = carry[(size_t)c * (B_DIM * D_DIM) + (b << 10) + d];
    size_t base = (size_t)(b * L_DIM + c * CHUNK) * D_DIM + d;
    for (int l = 0; l < CHUNK; ++l) {
        size_t ix = base + (size_t)l * D_DIM;
        s = fmaf(a, s, v[ix]);
        yb[ix] = f2bf(o[ix] * s);
    }
}

extern "C" void kernel_launch(void* const* d_in, const int* in_sizes, int n_in,
                              void* d_out, int out_size, void* d_ws, size_t ws_size,
                              hipStream_t stream) {
    const float* x     = (const float*)d_in[0];   // [B,L,D]
    const float* W_in  = (const float*)d_in[1];   // [3D,D]
    const float* Avec  = (const float*)d_in[2];   // [D]
    const float* W_out = (const float*)d_in[3];   // [D,D]
    float* out = (float*)d_out;                   // [B,L,D] f32

    char* ws = (char*)d_ws;
    size_t off = 0;
    auto alloc = [&](size_t bytes) -> char* {
        char* p = ws + off;
        off = (off + bytes + 255) & ~(size_t)255;
        return p;
    };
    unsigned short* xb  = (unsigned short*)alloc((size_t)M_TOT * D_DIM * 2);
    unsigned short* wib = (unsigned short*)alloc((size_t)3 * D_DIM * D_DIM * 2);
    unsigned short* wob = (unsigned short*)alloc((size_t)D_DIM * D_DIM * 2);
    unsigned short* yb  = (unsigned short*)alloc((size_t)M_TOT * D_DIM * 2);
    float* vbuf = (float*)alloc((size_t)M_TOT * D_DIM * 4);
    float* obuf = (float*)alloc((size_t)M_TOT * D_DIM * 4);
    float* rbuf = (float*)alloc((size_t)NCH * B_DIM * D_DIM * 4);
    float* cbuf = (float*)alloc((size_t)NCH * B_DIM * D_DIM * 4);

    // casts to bf16
    cvt2_kernel<<<4096, 256, 0, stream>>>((const float2*)x, (unsigned*)xb, (M_TOT * D_DIM) / 2);
    cvt2_kernel<<<1024, 256, 0, stream>>>((const float2*)W_in, (unsigned*)wib, (3 * D_DIM * D_DIM) / 2);
    cvt2_kernel<<<512, 256, 0, stream>>>((const float2*)W_out, (unsigned*)wob, (D_DIM * D_DIM) / 2);

    dim3 g1(M_TOT / MTILE, D_DIM / NTILE);   // 256 x 16
    // fused: proj (u,i,o) + activations -> v, o
    gemm_bf16_kernel<3><<<g1, 256, 0, stream>>>(xb, wib, vbuf, obuf);

    scan_pass1<<<(NCH * B_DIM * D_DIM) / 256, 256, 0, stream>>>(vbuf, Avec, rbuf);
    scan_pass2<<<(B_DIM * D_DIM) / 256, 256, 0, stream>>>(rbuf, Avec, cbuf);
    scan_pass3<<<(NCH * B_DIM * D_DIM) / 256, 256, 0, stream>>>(vbuf, obuf, Avec, cbuf, yb);

    // out = y @ W_out^T
    gemm_bf16_kernel<1><<<g1, 256, 0, stream>>>(yb, wob, out, nullptr);
}
